// PointNetFPModule_71691594104934
// MI455X (gfx1250) — matrix operand order repaired and checked by reading, not compiled, not run on hardware
//
#include <hip/hip_runtime.h>

// ---------------- problem constants (match reference setup_inputs) ----------
#define B_   16
#define N1_  4096
#define N2_  1024
#define C1_  256
#define C2_  512
#define CIN_ 768            // C2 + C1
#define CH_  256            // mlp[0]
#define COUT_ 256           // mlp[1]
#define M_   (B_ * N1_)     // 65536 rows

// ---------------- bf16 helpers ---------------------------------------------
typedef __attribute__((ext_vector_type(16))) __bf16 v16bf;
typedef __attribute__((ext_vector_type(8)))  float  v8f;

static __device__ __forceinline__ unsigned short f2bf(float f) {
    unsigned int u = __float_as_uint(f);
    unsigned int r = 0x7FFFu + ((u >> 16) & 1u);   // round to nearest even
    return (unsigned short)((u + r) >> 16);
}

union Frag32 { uint4 u[2]; v16bf v; };             // 32 bytes = 16 bf16 = 8 VGPRs

// ---------------- kernel 1: three_nn + inverse-distance weights -------------
// one block (256 thr) handles 256 query points of one batch; xyz2[b] in LDS
__global__ __launch_bounds__(256)
void three_nn_kernel(const float* __restrict__ xyz1, const float* __restrict__ xyz2,
                     float* __restrict__ w3, int* __restrict__ idx3)
{
    __shared__ float s2[N2_ * 3];                   // 12 KB
    const int b    = blockIdx.x / (N1_ / 256);
    const int tile = blockIdx.x % (N1_ / 256);
    const float* src = xyz2 + (size_t)b * N2_ * 3;
    for (int i = threadIdx.x; i < N2_ * 3; i += 256) s2[i] = src[i];
    __syncthreads();

    const int n = tile * 256 + threadIdx.x;
    const size_t p = (size_t)b * N1_ + n;
    const float px = xyz1[p * 3 + 0];
    const float py = xyz1[p * 3 + 1];
    const float pz = xyz1[p * 3 + 2];

    float d0 = 3.4e38f, d1 = 3.4e38f, d2 = 3.4e38f;
    int   i0 = 0, i1 = 0, i2 = 0;
    for (int j = 0; j < N2_; ++j) {
        const float dx = px - s2[j * 3 + 0];
        const float dy = py - s2[j * 3 + 1];
        const float dz = pz - s2[j * 3 + 2];
        const float d  = dx * dx + dy * dy + dz * dz;
        if (d < d0)      { d2 = d1; i2 = i1; d1 = d0; i1 = i0; d0 = d; i0 = j; }
        else if (d < d1) { d2 = d1; i2 = i1; d1 = d;  i1 = j; }
        else if (d < d2) { d2 = d;  i2 = j; }
    }
    d0 = fmaxf(d0, 1e-10f); d1 = fmaxf(d1, 1e-10f); d2 = fmaxf(d2, 1e-10f);
    const float v0 = 1.f / d0, v1 = 1.f / d1, v2 = 1.f / d2;
    const float inv = 1.f / (v0 + v1 + v2);
    w3[p * 3 + 0] = v0 * inv;  idx3[p * 3 + 0] = i0;
    w3[p * 3 + 1] = v1 * inv;  idx3[p * 3 + 1] = i1;
    w3[p * 3 + 2] = v2 * inv;  idx3[p * 3 + 2] = i2;
}

// ---------------- kernel 2: transpose + bf16-pack both weight matrices ------
__global__ __launch_bounds__(256)
void pack_weights(const float* __restrict__ W1, const float* __restrict__ W2,
                  unsigned short* __restrict__ W1t, unsigned short* __restrict__ W2t)
{
    const int g = blockIdx.x * 256 + threadIdx.x;
    const int NW1 = CIN_ * CH_;                    // 196608
    if (g < NW1) {
        const int n = g / CIN_, k = g % CIN_;      // W1t[n][k] = W1[k][n]
        W1t[(size_t)n * CIN_ + k] = f2bf(W1[(size_t)k * CH_ + n]);
    } else if (g < NW1 + CH_ * COUT_) {
        const int g2 = g - NW1;
        const int n = g2 / CH_, k = g2 % CH_;      // W2t[n][k] = W2[k][n]
        W2t[(size_t)n * CH_ + k] = f2bf(W2[(size_t)k * COUT_ + n]);
    }
}

// ---------------- kernel 3: three_interpolate + concat -> X bf16 [M][768] ---
// one block per query point
__global__ __launch_bounds__(256)
void interp_pack(const float* __restrict__ points1, const float* __restrict__ points2,
                 const float* __restrict__ w3, const int* __restrict__ idx3,
                 unsigned short* __restrict__ X)
{
    const size_t p = blockIdx.x;                   // 0 .. M_-1
    const int b = (int)(p / N1_);
    const float w0 = w3[p * 3 + 0], w1 = w3[p * 3 + 1], w2 = w3[p * 3 + 2];
    const float* r0 = points2 + ((size_t)b * N2_ + idx3[p * 3 + 0]) * C2_;
    const float* r1 = points2 + ((size_t)b * N2_ + idx3[p * 3 + 1]) * C2_;
    const float* r2 = points2 + ((size_t)b * N2_ + idx3[p * 3 + 2]) * C2_;
    unsigned short* xr = X + p * CIN_;
    for (int c = threadIdx.x; c < C2_; c += 256)
        xr[c] = f2bf(w0 * r0[c] + w1 * r1[c] + w2 * r2[c]);
    const float* p1 = points1 + p * C1_;
    for (int c = threadIdx.x; c < C1_; c += 256)
        xr[C2_ + c] = f2bf(p1[c]);
}

// ---------------- kernel 4/5: bf16 WMMA GEMM + bias + ReLU ------------------
// X [M][K] bf16 row-major, Wt [N][K] bf16 (pre-transposed), Y [M][N].
// Wave tile 32(M) x 128(N): 16 accumulator tiles. Block = 4 waves -> 128x128.
template <int K, int N, bool OUT_BF16>
__global__ __launch_bounds__(128)
void gemm_bias_relu(const unsigned short* __restrict__ X,
                    const unsigned short* __restrict__ Wt,
                    const float* __restrict__ bias,
                    void* __restrict__ Yout)
{
    const int lane = threadIdx.x & 31;
    const int wave = threadIdx.x >> 5;
    const int m0   = blockIdx.x * 128 + wave * 32;
    const int n0   = blockIdx.y * 128;
    const int half = lane >> 4;                    // 0: lanes 0-15, 1: lanes 16-31
    const int l15  = lane & 15;

    v8f acc[2][8];
#pragma unroll
    for (int mt = 0; mt < 2; ++mt)
#pragma unroll
        for (int nt = 0; nt < 8; ++nt) acc[mt][nt] = v8f{};

    const unsigned short* Arow0 = X + (size_t)(m0 + l15) * K;
    const unsigned short* Arow1 = X + (size_t)(m0 + 16 + l15) * K;

    for (int kb = 0; kb < K; kb += 32) {
        // A fragments per ISA 16-bit A layout: lanes 0-15 hold K {0..7,16..23},
        // lanes 16-31 hold K {8..15,24..31} of their row.
        Frag32 a0, a1;
        a0.u[0] = *(const uint4*)(Arow0 + kb + 8 * half);
        a0.u[1] = *(const uint4*)(Arow0 + kb + 16 + 8 * half);
        a1.u[0] = *(const uint4*)(Arow1 + kb + 8 * half);
        a1.u[1] = *(const uint4*)(Arow1 + kb + 16 + 8 * half);
#pragma unroll
        for (int nt = 0; nt < 8; ++nt) {
            // B fragment: col = l15, lanes 0-15 K {0..7,16..23}, lanes 16-31 K {8..15,24..31}
            const unsigned short* Brow =
                Wt + (size_t)(n0 + nt * 16 + l15) * K + kb + 8 * half;
            Frag32 bf;
            bf.u[0] = *(const uint4*)(Brow);
            bf.u[1] = *(const uint4*)(Brow + 16);
            acc[0][nt] = __builtin_amdgcn_wmma_f32_16x16x32_bf16(
                false, a0.v, false, bf.v, (short)0, acc[0][nt], false, false);
            acc[1][nt] = __builtin_amdgcn_wmma_f32_16x16x32_bf16(
                false, a1.v, false, bf.v, (short)0, acc[1][nt], false, false);
        }
    }

    // epilogue: C/D layout -> VGPR r: lanes 0-15 M=r, lanes 16-31 M=r+8; N=l15
#pragma unroll
    for (int nt = 0; nt < 8; ++nt) {
        const int col = n0 + nt * 16 + l15;
        const float bv = bias[col];
#pragma unroll
        for (int mt = 0; mt < 2; ++mt) {
#pragma unroll
            for (int r = 0; r < 8; ++r) {
                const int row = m0 + mt * 16 + r + 8 * half;
                float v = acc[mt][nt][r] + bv;
                v = v > 0.f ? v : 0.f;
                if (OUT_BF16)
                    ((unsigned short*)Yout)[(size_t)row * N + col] = f2bf(v);
                else
                    ((float*)Yout)[(size_t)row * N + col] = v;
            }
        }
    }
}

// ---------------- launch ----------------------------------------------------
extern "C" void kernel_launch(void* const* d_in, const int* in_sizes, int n_in,
                              void* d_out, int out_size, void* d_ws, size_t ws_size,
                              hipStream_t stream)
{
    (void)in_sizes; (void)n_in; (void)out_size; (void)ws_size;
    const float* xyz1    = (const float*)d_in[0];
    const float* xyz2    = (const float*)d_in[1];
    const float* points1 = (const float*)d_in[2];
    const float* points2 = (const float*)d_in[3];
    const float* W1      = (const float*)d_in[4];
    const float* b1      = (const float*)d_in[5];
    const float* W2      = (const float*)d_in[6];
    const float* b2      = (const float*)d_in[7];

    // workspace layout (256B aligned)
    char* ws = (char*)d_ws;
    size_t off = 0;
    float* w3   = (float*)(ws + off);          off += (size_t)M_ * 3 * sizeof(float);   // 786 KB
    off = (off + 255) & ~(size_t)255;
    int* idx3   = (int*)(ws + off);            off += (size_t)M_ * 3 * sizeof(int);     // 786 KB
    off = (off + 255) & ~(size_t)255;
    unsigned short* Xb  = (unsigned short*)(ws + off); off += (size_t)M_ * CIN_ * 2;    // 100.7 MB
    off = (off + 255) & ~(size_t)255;
    unsigned short* W1t = (unsigned short*)(ws + off); off += (size_t)CIN_ * CH_ * 2;   // 393 KB
    off = (off + 255) & ~(size_t)255;
    unsigned short* W2t = (unsigned short*)(ws + off); off += (size_t)CH_ * COUT_ * 2;  // 131 KB
    off = (off + 255) & ~(size_t)255;
    unsigned short* Hb  = (unsigned short*)(ws + off); off += (size_t)M_ * CH_ * 2;     // 33.5 MB

    // 1) three_nn + weights
    three_nn_kernel<<<B_ * (N1_ / 256), 256, 0, stream>>>(xyz1, xyz2, w3, idx3);

    // 2) transpose + bf16-pack W1, W2
    {
        const int total = CIN_ * CH_ + CH_ * COUT_;
        pack_weights<<<(total + 255) / 256, 256, 0, stream>>>(W1, W2, W1t, W2t);
    }

    // 3) interpolate + concat -> X bf16
    interp_pack<<<M_, 256, 0, stream>>>(points1, points2, w3, idx3, Xb);

    // 4) GEMM1: [M,768]x[768,256] + b1 + ReLU -> H bf16
    gemm_bias_relu<CIN_, CH_, true><<<dim3(M_ / 128, CH_ / 128), 128, 0, stream>>>(
        Xb, W1t, b1, (void*)Hb);

    // 5) GEMM2: [M,256]x[256,256] + b2 + ReLU -> out fp32
    gemm_bias_relu<CH_, COUT_, false><<<dim3(M_ / 128, COUT_ / 128), 128, 0, stream>>>(
        Hb, W2t, b2, d_out);
}
